// HierarchicalPredictiveCoding_26336739459343
// MI455X (gfx1250) — matrix operand order repaired and checked by reading, not compiled, not run on hardware
//
#include <hip/hip_runtime.h>
#include <hip/hip_bf16.h>

// ---------------------------------------------------------------------------
// Types for CDNA5 WMMA / TDM (gfx1250, wave32)
// ---------------------------------------------------------------------------
typedef __attribute__((ext_vector_type(16))) __bf16 v16bf;
typedef __attribute__((ext_vector_type(8)))  __bf16 v8bf;
typedef __attribute__((ext_vector_type(4)))  __bf16 v4bf;
typedef __attribute__((ext_vector_type(8)))  float  v8f;
typedef __attribute__((ext_vector_type(4)))  unsigned int u32x4;
typedef __attribute__((ext_vector_type(8)))  int i32x8;
typedef __attribute__((ext_vector_type(4)))  int i32x4;

#define BM 256
#define BN 128
#define BK 32
#define LPAD 8            // LDS row pad (elems) == TDM pad: 4 dwords per 16 dwords
#define AROW (BK + LPAD)  // padded LDS row stride (elements)

#if __has_builtin(__builtin_amdgcn_tensor_load_to_lds)
#define HAVE_TDM 1
#else
#define HAVE_TDM 0
#endif

// ---------------------------------------------------------------------------
// f32 -> bf16 elementwise convert (4-wide, grid-stride). n must be /4.
// ---------------------------------------------------------------------------
__global__ void f32_to_bf16_kernel(const float* __restrict__ in,
                                   __bf16* __restrict__ out, long n4) {
  long i = (long)blockIdx.x * blockDim.x + threadIdx.x;
  long stride = (long)gridDim.x * blockDim.x;
  const float4* in4 = (const float4*)in;
  v4bf* out4 = (v4bf*)out;
  for (; i < n4; i += stride) {
    float4 v = in4[i];
    v4bf o;
    o.x = (__bf16)v.x; o.y = (__bf16)v.y; o.z = (__bf16)v.z; o.w = (__bf16)v.w;
    out4[i] = o;
  }
}

// ---------------------------------------------------------------------------
// W[K,N] f32 -> Wt[N,K] bf16   (tiled transpose, K,N multiples of 32)
// ---------------------------------------------------------------------------
__global__ __launch_bounds__(1024)
void transpose_bf16_kernel(const float* __restrict__ W, __bf16* __restrict__ Wt,
                           int K, int N) {
  __shared__ float tile[32][33];
  int k = blockIdx.y * 32 + threadIdx.y;
  int n = blockIdx.x * 32 + threadIdx.x;
  tile[threadIdx.y][threadIdx.x] = W[(size_t)k * N + n];
  __syncthreads();
  int n2 = blockIdx.x * 32 + threadIdx.y;
  int k2 = blockIdx.y * 32 + threadIdx.x;
  Wt[(size_t)n2 * K + k2] = (__bf16)tile[threadIdx.x][threadIdx.y];
}

// ---------------------------------------------------------------------------
// prec2 = softplus(log_prec)^2
// ---------------------------------------------------------------------------
__global__ void softplus2_kernel(const float* __restrict__ in,
                                 float* __restrict__ out, int n) {
  int i = blockIdx.x * blockDim.x + threadIdx.x;
  if (i < n) {
    float x = in[i];
    float sp = (x > 20.0f) ? x : log1pf(expf(x));
    out[i] = sp * sp;
  }
}

// ---------------------------------------------------------------------------
// Row LayerNorm: outf = LN(T)*g + b  (optionally also bf16 copy)
// ---------------------------------------------------------------------------
__global__ __launch_bounds__(256)
void layernorm_kernel(const float* __restrict__ T, const float* __restrict__ g,
                      const float* __restrict__ b, float* __restrict__ outf,
                      __bf16* __restrict__ outb, int N) {
  const int row = blockIdx.x;
  const float* rp = T + (size_t)row * N;
  float s = 0.f, ss = 0.f;
  for (int c = threadIdx.x; c < N; c += blockDim.x) {
    float v = rp[c];
    s += v; ss += v * v;
  }
  for (int o = 16; o > 0; o >>= 1) {
    s  += __shfl_xor(s, o, 32);
    ss += __shfl_xor(ss, o, 32);
  }
  __shared__ float red[2][8];
  int wid = threadIdx.x >> 5, lid = threadIdx.x & 31;
  if (lid == 0) { red[0][wid] = s; red[1][wid] = ss; }
  __syncthreads();
  if (wid == 0) {
    s  = (lid < 8) ? red[0][lid] : 0.f;
    ss = (lid < 8) ? red[1][lid] : 0.f;
    for (int o = 4; o > 0; o >>= 1) {
      s  += __shfl_xor(s, o, 32);
      ss += __shfl_xor(ss, o, 32);
    }
    if (lid == 0) { red[0][0] = s; red[1][0] = ss; }
  }
  __syncthreads();
  float inv = 1.0f / (float)N;
  float mu  = red[0][0] * inv;
  float var = red[1][0] * inv - mu * mu;
  float rs  = rsqrtf(var + 1e-5f);
  float* op = outf + (size_t)row * N;
  __bf16* ob = outb ? outb + (size_t)row * N : nullptr;
  for (int c = threadIdx.x; c < N; c += blockDim.x) {
    float y = (rp[c] - mu) * rs * g[c] + b[c];
    op[c] = y;
    if (ob) ob[c] = (__bf16)y;
  }
}

#if HAVE_TDM
// ---------------------------------------------------------------------------
// Issue one TDM tile load: rows x BK bf16 tile from row-major [.,K] matrix
// into LDS laid out as [rows][AROW] (pad: 4 dwords after every 16 dwords).
// D# layout per CDNA5 ISA 8.3/8.4. Issued once per calling wave (EXEC-indep).
// ---------------------------------------------------------------------------
__device__ __forceinline__ void tdm_load_tile(unsigned long long gaddr,
                                              unsigned int lds_off,
                                              int K, int rows) {
  u32x4 g0;
  g0.x = 1u;                                   // count=1 (valid descriptor)
  g0.y = lds_off;                              // lds_addr
  g0.z = (unsigned int)gaddr;                  // global_addr[31:0]
  g0.w = (unsigned int)((gaddr >> 32) & 0x01FFFFFFu) | 0x80000000u; // addr[56:32] | type=2
  i32x8 g1;
  // data_size=1 (2B) <<16 | pad_enable<<20 | pad_interval=3 (16 dwords) <<22
  // | pad_amount=3 (4 dwords) <<25
  g1[0] = (1 << 16) | (1 << 20) | (3 << 22) | (3 << 25);
  g1[1] = (K & 0xFFFF) << 16;                             // tensor_dim0[15:0]
  g1[2] = ((K >> 16) & 0xFFFF) | ((rows & 0xFFFF) << 16); // tensor_dim0[31:16] | tensor_dim1[15:0]
  g1[3] = (BK << 16);                                     // tensor_dim1[31:16]=0 | tile_dim0=BK
  g1[4] = rows & 0xFFFF;                                  // tile_dim1 | tile_dim2=0
  g1[5] = K;                                              // tensor_dim0_stride[31:0]
  g1[6] = 0;                                              // stride0[47:32] | stride1[15:0]
  g1[7] = 0;
  i32x4 gz = (i32x4){0, 0, 0, 0};
#if defined(__clang_major__) && (__clang_major__ >= 23)
  i32x8 gz8 = (i32x8){0, 0, 0, 0, 0, 0, 0, 0};
  __builtin_amdgcn_tensor_load_to_lds(g0, g1, gz, gz, gz8, 0);
#else
  __builtin_amdgcn_tensor_load_to_lds(g0, g1, gz, gz, 0);
#endif
}

__device__ __forceinline__ void tdm_wait0() {
#if __has_builtin(__builtin_amdgcn_s_wait_tensorcnt)
  __builtin_amdgcn_s_wait_tensorcnt(0);
#else
  asm volatile("s_wait_tensorcnt 0x0" ::: "memory");
#endif
}
#endif // HAVE_TDM

// ---------------------------------------------------------------------------
// NT bf16 GEMM with fused epilogue:  acc = A[M,K] * B[N,K]^T   (f32 accum)
//   EPI==0 : outf[m,n] = acc + bias[n]
//   EPI==1 : outb[m,n] = bf16( (xref[m,n] - (acc + bias[n])) * prec2[n] )
//   EPI==2 : outf[m,n] = hprev[m,n] + 0.2f * acc
// Block tile 256x128, 8 wave32 waves, each wave a 64x64 sub-tile
// (4x4 grid of v_wmma_f32_16x16x32_bf16 accumulators). Tiles staged to LDS
// by the Tensor Data Mover with double-buffered ping-pong so the DMA runs
// concurrently with the WMMA pipes.
// M multiple of 256, N multiple of 128, K multiple of 32.
// ---------------------------------------------------------------------------
template <int EPI>
__global__ __launch_bounds__(256)
void gemm_bf16_nt_kernel(const __bf16* __restrict__ A, const __bf16* __restrict__ Bm,
                         int M, int N, int K,
                         const float* __restrict__ bias,
                         const float* __restrict__ xref,
                         const float* __restrict__ prec2,
                         const float* __restrict__ hprev,
                         float* __restrict__ outf,
                         __bf16* __restrict__ outb) {
  __shared__ __align__(16) __bf16 As[2][BM][AROW];   // 2 x 20 KB
  __shared__ __align__(16) __bf16 Bs[2][BN][AROW];   // 2 x 10 KB

  const int tid  = threadIdx.x;
  const int lane = tid & 31;
  const int wave = tid >> 5;       // 0..7
  const int wm   = wave & 3;       // 4 waves along M (64 each)
  const int wn   = wave >> 2;      // 2 waves along N (64 each)
  const int mbase = blockIdx.y * BM;
  const int nbase = blockIdx.x * BN;

  v8f acc[4][4];
#pragma unroll
  for (int i = 0; i < 4; ++i)
#pragma unroll
    for (int j = 0; j < 4; ++j)
      acc[i][j] = (v8f){0.f, 0.f, 0.f, 0.f, 0.f, 0.f, 0.f, 0.f};

  const int hi  = lane >> 4;          // lane half
  const int ml  = lane & 15;
  const int ac0 = hi * 8;             // A: K chunks at ac0 and ac0+16
  const int bk0 = hi * 8;             // B: K chunks at bk0 and bk0+16

  const int nk = K / BK;

  // compute one staged K-tile from LDS buffer `cur`
  auto compute_tile = [&](int cur) {
    v16bf afrag[4];
#pragma unroll
    for (int i = 0; i < 4; ++i) {
      int r = wm * 64 + i * 16 + ml;
      v8bf lo  = *(const v8bf*)&As[cur][r][ac0];
      v8bf hi8 = *(const v8bf*)&As[cur][r][ac0 + 16];
      afrag[i] = __builtin_shufflevector(lo, hi8, 0, 1, 2, 3, 4, 5, 6, 7,
                                                  8, 9, 10, 11, 12, 13, 14, 15);
    }
#pragma unroll
    for (int j = 0; j < 4; ++j) {
      int c = wn * 64 + j * 16 + ml;
      v8bf lo  = *(const v8bf*)&Bs[cur][c][bk0];
      v8bf hi8 = *(const v8bf*)&Bs[cur][c][bk0 + 16];
      v16bf bfrag = __builtin_shufflevector(lo, hi8, 0, 1, 2, 3, 4, 5, 6, 7,
                                                     8, 9, 10, 11, 12, 13, 14, 15);
#pragma unroll
      for (int i = 0; i < 4; ++i)
        acc[i][j] = __builtin_amdgcn_wmma_f32_16x16x32_bf16(
            false, afrag[i], false, bfrag, (short)0, acc[i][j], false, false);
    }
  };

#if HAVE_TDM
  // prologue: fill buffer 0 for kt=0
  if (wave == 0) {
    unsigned long long ga = (unsigned long long)(A  + (size_t)mbase * K);
    unsigned long long gb = (unsigned long long)(Bm + (size_t)nbase * K);
    tdm_load_tile(ga, (unsigned int)(unsigned long long)&As[0][0][0], K, BM);
    tdm_load_tile(gb, (unsigned int)(unsigned long long)&Bs[0][0][0], K, BN);
  }
  for (int kt = 0; kt < nk; ++kt) {
    const int cur = kt & 1;
    if (wave == 0) tdm_wait0();          // current buffer's pair complete
    __syncthreads();                     // everyone sees buffer `cur`
    if (wave == 0 && kt + 1 < nk) {      // start DMA of next tile into other buffer
      const int nxt = cur ^ 1;
      unsigned long long ga =
          (unsigned long long)(A  + (size_t)mbase * K + (size_t)(kt + 1) * BK);
      unsigned long long gb =
          (unsigned long long)(Bm + (size_t)nbase * K + (size_t)(kt + 1) * BK);
      tdm_load_tile(ga, (unsigned int)(unsigned long long)&As[nxt][0][0], K, BM);
      tdm_load_tile(gb, (unsigned int)(unsigned long long)&Bs[nxt][0][0], K, BN);
    }
    compute_tile(cur);                   // 16 WMMAs per wave, overlapped with DMA
    __syncthreads();                     // compute done before `cur` is refilled at kt+2
  }
#else
  for (int kt = 0; kt < nk; ++kt) {
    // manual staging fallback (single buffer 0)
    const __bf16* ga = A + (size_t)(mbase + tid) * K + (size_t)kt * BK;
    const int brow = tid >> 1;
    const int bcol = (tid & 1) * 16;
    const __bf16* gb = Bm + (size_t)(nbase + brow) * K + (size_t)kt * BK + bcol;
    if (kt + 1 < nk) {
      __builtin_prefetch(ga + BK, 0, 1);
      __builtin_prefetch(gb + BK, 0, 1);
    }
    v8bf a0 = *(const v8bf*)ga;
    v8bf a1 = *(const v8bf*)(ga + 8);
    v8bf a2 = *(const v8bf*)(ga + 16);
    v8bf a3 = *(const v8bf*)(ga + 24);
    v8bf b0 = *(const v8bf*)gb;
    v8bf b1 = *(const v8bf*)(gb + 8);
    *(v8bf*)&As[0][tid][0]  = a0;
    *(v8bf*)&As[0][tid][8]  = a1;
    *(v8bf*)&As[0][tid][16] = a2;
    *(v8bf*)&As[0][tid][24] = a3;
    *(v8bf*)&Bs[0][brow][bcol]     = b0;
    *(v8bf*)&Bs[0][brow][bcol + 8] = b1;
    __syncthreads();
    compute_tile(0);
    __syncthreads();
  }
#endif

  // Epilogue. C layout: VGPR r of lane l -> M = r + 8*(l>=16), N = l&15.
#pragma unroll
  for (int i = 0; i < 4; ++i) {
#pragma unroll
    for (int j = 0; j < 4; ++j) {
      int mt = mbase + wm * 64 + i * 16 + hi * 8;
      int nn = nbase + wn * 64 + j * 16 + ml;
#pragma unroll
      for (int r = 0; r < 8; ++r) {
        size_t idx = (size_t)(mt + r) * N + nn;
        float v = acc[i][j][r];
        if (EPI == 0) {
          outf[idx] = v + bias[nn];
        } else if (EPI == 1) {
          float e = (xref[idx] - (v + bias[nn])) * prec2[nn];
          outb[idx] = (__bf16)e;
        } else {
          outf[idx] = hprev[idx] + 0.2f * v;
        }
      }
    }
  }
}

// ---------------------------------------------------------------------------
// Host orchestration
// ---------------------------------------------------------------------------
static inline void* ws_take(char*& p, size_t bytes) {
  void* r = (void*)p;
  p += (bytes + 255) & ~(size_t)255;
  return r;
}

static void launch_gemm(int epi, const __bf16* A, const __bf16* B,
                        int M, int N, int K,
                        const float* bias, const float* xref, const float* prec2,
                        const float* hprev, float* outf, __bf16* outb,
                        hipStream_t stream) {
  dim3 grid(N / BN, M / BM);
  dim3 block(256);
  if (epi == 0)
    gemm_bf16_nt_kernel<0><<<grid, block, 0, stream>>>(A, B, M, N, K, bias, xref, prec2, hprev, outf, outb);
  else if (epi == 1)
    gemm_bf16_nt_kernel<1><<<grid, block, 0, stream>>>(A, B, M, N, K, bias, xref, prec2, hprev, outf, outb);
  else
    gemm_bf16_nt_kernel<2><<<grid, block, 0, stream>>>(A, B, M, N, K, bias, xref, prec2, hprev, outf, outb);
}

static void launch_cvt(const float* in, __bf16* out, long n, hipStream_t stream) {
  long n4 = n / 4;
  int blocks = (int)((n4 + 255) / 256);
  if (blocks > 4096) blocks = 4096;
  f32_to_bf16_kernel<<<blocks, 256, 0, stream>>>(in, out, n4);
}

extern "C" void kernel_launch(void* const* d_in, const int* in_sizes, int n_in,
                              void* d_out, int out_size, void* d_ws, size_t ws_size,
                              hipStream_t stream) {
  (void)in_sizes; (void)n_in; (void)out_size; (void)ws_size;
  const int B = 8192;
  const int DIM0 = 4096, DIM1 = 2048, DIM2 = 1024;

  const float* x = (const float*)d_in[0];

  char* wp = (char*)d_ws;
  __bf16* XB  = (__bf16*)ws_take(wp, (size_t)B * DIM0 * 2);
  __bf16* WR  = (__bf16*)ws_take(wp, (size_t)DIM1 * DIM0 * 2);
  __bf16* WG  = (__bf16*)ws_take(wp, (size_t)DIM0 * DIM1 * 2);
  __bf16* WGT = (__bf16*)ws_take(wp, (size_t)DIM1 * DIM0 * 2);
  float*  P2  = (float*) ws_take(wp, (size_t)DIM0 * 4);
  float*  H0  = (float*) ws_take(wp, (size_t)B * DIM1 * 4);
  __bf16* HB0 = (__bf16*)ws_take(wp, (size_t)B * DIM1 * 2);
  float*  H1  = (float*) ws_take(wp, (size_t)B * DIM2 * 4);
  __bf16* HB1 = (__bf16*)ws_take(wp, (size_t)B * DIM2 * 2);
  float*  T   = (float*) ws_take(wp, (size_t)B * DIM1 * 4);
  __bf16* EB  = (__bf16*)ws_take(wp, (size_t)B * DIM0 * 2);

  launch_cvt(x, XB, (long)B * DIM0, stream);

  for (int li = 0; li < 2; ++li) {
    const int ind = (li == 0) ? DIM0 : DIM1;
    const int hid = (li == 0) ? DIM1 : DIM2;
    const int base = 1 + 7 * li;
    const float* gen_w    = (const float*)d_in[base + 0];
    const float* gen_b    = (const float*)d_in[base + 1];
    const float* rec_w    = (const float*)d_in[base + 2];
    const float* rec_b    = (const float*)d_in[base + 3];
    const float* log_prec = (const float*)d_in[base + 4];
    const float* ln_g     = (const float*)d_in[base + 5];
    const float* ln_b     = (const float*)d_in[base + 6];

    const float*  xf = (li == 0) ? x  : H0;
    const __bf16* xb = (li == 0) ? XB : HB0;
    float*  Hf = (li == 0) ? H0  : H1;
    __bf16* Hb = (li == 0) ? HB0 : HB1;

    launch_cvt(rec_w, WR, (long)hid * ind, stream);
    launch_cvt(gen_w, WG, (long)ind * hid, stream);
    {
      dim3 g(hid / 32, ind / 32), blk(32, 32);
      transpose_bf16_kernel<<<g, blk, 0, stream>>>(gen_w, WGT, ind, hid);
    }
    softplus2_kernel<<<(ind + 255) / 256, 256, 0, stream>>>(log_prec, P2, ind);

    // h = LN(x @ rec_w^T + rec_b)
    launch_gemm(0, xb, WR, B, hid, ind, rec_b, nullptr, nullptr, nullptr, T, nullptr, stream);
    layernorm_kernel<<<B, 256, 0, stream>>>(T, ln_g, ln_b, Hf, Hb, hid);

    for (int it = 0; it < 3; ++it) {
      // err = (x - (h @ gen_w^T + gen_b)) * prec2  -> bf16
      launch_gemm(1, Hb, WG, B, ind, hid, gen_b, xf, P2, nullptr, nullptr, EB, stream);
      // t = h + 0.2 * (err @ gen_w)
      launch_gemm(2, EB, WGT, B, hid, ind, nullptr, nullptr, nullptr, Hf, T, nullptr, stream);
      // h = LN(t)
      bool last = (li == 1) && (it == 2);
      float*  of = last ? (float*)d_out : Hf;
      __bf16* ob = last ? nullptr : Hb;
      layernorm_kernel<<<B, 256, 0, stream>>>(T, ln_g, ln_b, of, ob, hid);
    }
  }
}
